// AttentionOp_5506148073832
// MI455X (gfx1250) — matrix operand order repaired and checked by reading, not compiled, not run on hardware
//
#include <hip/hip_runtime.h>
#include <hip/hip_bf16.h>
#include <math.h>

typedef __attribute__((ext_vector_type(16))) __bf16 v16bf;
typedef __attribute__((ext_vector_type(8)))  float  v8f;

namespace {
constexpr int B_   = 2;
constexpr int T_   = 2048;
constexpr int NH   = 16;
constexpr int NKV  = 8;
constexpr int D_   = 128;
constexpr int G_   = NH / NKV;
constexpr int BQ   = 64;     // query rows per workgroup (4 waves x 16)
constexpr int STILE = 32;    // keys per s-tile iteration
constexpr int VSTR = 40;     // LDS V^T row stride (bf16): 80B rows, 16B aligned, 20-bank stride
constexpr int KSTR = 136;    // LDS K row stride (bf16): 272B rows, 16B aligned, 4-bank lane stride
constexpr float MASKVAL = -0.7f * 3.4028234663852886e38f;
constexpr float LOG2E_E = 1.4426940729f;  // log2(2.71828)
}

union BF16Frag { v16bf v; unsigned short u[16]; uint4 q[2]; };

__device__ __forceinline__ unsigned short f2bf(float f) {
  return (unsigned short)(__float_as_uint(f) >> 16);
}

__device__ __forceinline__ v8f wmma_bf16(v16bf a, v16bf b, v8f c) {
  // D = A(16x32 bf16) * B(32x16 bf16) + C(16x16 f32)
  return __builtin_amdgcn_wmma_f32_16x16x32_bf16(false, a, false, b, (short)0, c,
                                                 false, false);
}

__global__ __launch_bounds__(128, 1) void fa_fwd_kernel(
    const float* __restrict__ Q, const float* __restrict__ K,
    const float* __restrict__ V, const int* __restrict__ SEG,
    float* __restrict__ outO, float* __restrict__ outM, float* __restrict__ outL) {
  __shared__ unsigned short Kt[STILE * KSTR];  // K tile as bf16: [key][d], padded
  __shared__ unsigned short Vt[D_ * VSTR];     // V^T tile as bf16: [dcol][key], padded
  __shared__ int segk[STILE];

  const int tid  = threadIdx.x;
  const int lane = tid & 31;
  const int wave = tid >> 5;
  const int col  = lane & 15;   // query column owned by this lane (S^T layout)
  const int hi   = lane >> 4;   // half-wave select

  int wg = blockIdx.x;
  const int qt = wg % (T_ / BQ); wg /= (T_ / BQ);
  const int h  = wg % NH;        wg /= NH;
  const int b  = wg;
  const int kvh = h / G_;
  const int wq0 = qt * BQ;          // workgroup query base
  const int q0  = wq0 + wave * 16;  // this wave's query base

  const float* Qb = Q + (((size_t)b * T_) * NH + h) * D_;
  const float* Kb = K + (((size_t)b * T_) * NKV + kvh) * D_;
  const float* Vb = V + (((size_t)b * T_) * NKV + kvh) * D_;
  const int*   Sb = SEG + (size_t)b * T_;

  // Staging assignments (uniform across the loop)
  const int kkey = tid >> 2;          // K staging: key row 0..31
  const int kd0  = (tid & 3) * 32;    // K staging: 32 consecutive d elems
  const int vkg  = tid >> 4;          // V staging: key group (4 keys) 0..7
  const int vd0  = (tid & 15) * 8;    // V staging: 8 consecutive d elems

  // ---- Q fragments, B-matrix layout (col = query, k along d in AMD table order) ----
  v16bf qf[4];
  {
    const float* qrow = Qb + (size_t)(q0 + col) * NH * D_;
#pragma unroll
    for (int c = 0; c < 4; ++c) {
      BF16Frag t;
      const float* p = qrow + c * 32 + hi * 8;
#pragma unroll
      for (int j = 0; j < 16; ++j) t.u[j] = f2bf(p[j + (j & 8)]);
      qf[c] = t.v;
    }
  }
  const int segq = Sb[q0 + col];

  const v8f vzero = {0.f, 0.f, 0.f, 0.f, 0.f, 0.f, 0.f, 0.f};
  v8f acc[8];
#pragma unroll
  for (int i = 0; i < 8; ++i) acc[i] = vzero;

  float m_run = -INFINITY;
  float l_run = 0.0f;

  const int ntiles = (wq0 / STILE) + 2;  // cover keys up to wq0+63 (causal)

  for (int it = 0; it < ntiles; ++it) {
    const int s0 = it * STILE;
    __syncthreads();

    // ---- stage K tile (bf16, row-major, padded): 1 key-quarter per thread ----
    {
      const float* p = Kb + (size_t)(s0 + kkey) * NKV * D_ + kd0;
      union { unsigned short u[32]; uint4 q[4]; } t;
#pragma unroll
      for (int j = 0; j < 32; ++j) t.u[j] = f2bf(p[j]);
      uint4* dst = (uint4*)&Kt[kkey * KSTR + kd0];
#pragma unroll
      for (int j = 0; j < 4; ++j) dst[j] = t.q[j];
    }

    // ---- stage V^T tile (bf16): 4 adjacent keys packed per b64 store ----
    {
      const float* p0 = Vb + (size_t)(s0 + vkg * 4 + 0) * NKV * D_ + vd0;
      const float* p1 = Vb + (size_t)(s0 + vkg * 4 + 1) * NKV * D_ + vd0;
      const float* p2 = Vb + (size_t)(s0 + vkg * 4 + 2) * NKV * D_ + vd0;
      const float* p3 = Vb + (size_t)(s0 + vkg * 4 + 3) * NKV * D_ + vd0;
#pragma unroll
      for (int j = 0; j < 8; ++j) {
        union { unsigned short u[4]; uint2 q; } w;
        w.u[0] = f2bf(p0[j]); w.u[1] = f2bf(p1[j]);
        w.u[2] = f2bf(p2[j]); w.u[3] = f2bf(p3[j]);
        *(uint2*)&Vt[(vd0 + j) * VSTR + vkg * 4] = w.q;
      }
    }
    if (tid < STILE) segk[tid] = Sb[s0 + tid];
    __syncthreads();

    // ---- prefetch next s-tile into L2/L0 while this tile computes ----
    if (it + 1 < ntiles) {
      const int s1 = s0 + STILE;
      __builtin_prefetch(Kb + (size_t)(s1 + kkey) * NKV * D_ + kd0, 0, 1);
      __builtin_prefetch(Vb + (size_t)(s1 + vkg * 4) * NKV * D_ + vd0, 0, 1);
    }

    // ---- S^T = K * Q^T : two 16x16 key blocks, accumulate over 4 d-chunks ----
    v8f sc0 = vzero, sc1 = vzero;
    const unsigned short* k0p = &Kt[col * KSTR];
    const unsigned short* k1p = &Kt[(16 + col) * KSTR];
#pragma unroll
    for (int c = 0; c < 4; ++c) {
      BF16Frag ka, kb;
      ka.q[0] = *(const uint4*)(k0p + c * 32 + hi * 8);
      ka.q[1] = *(const uint4*)(k0p + c * 32 + hi * 8 + 16);
      kb.q[0] = *(const uint4*)(k1p + c * 32 + hi * 8);
      kb.q[1] = *(const uint4*)(k1p + c * 32 + hi * 8 + 16);
      sc0 = wmma_bf16(ka.v, qf[c], sc0);
      sc1 = wmma_bf16(kb.v, qf[c], sc1);
    }

    // ---- mask + online softmax (lane owns query `col`; keys along regs) ----
    const int qidx = q0 + col;
    float sv[16];
#pragma unroll
    for (int r = 0; r < 8; ++r) {
      const int k0 = s0 + hi * 8 + r;
      const int k1 = k0 + 16;
      const bool ok0 = (k0 <= qidx) && (segk[hi * 8 + r] == segq);
      const bool ok1 = (k1 <= qidx) && (segk[16 + hi * 8 + r] == segq);
      sv[r]     = ok0 ? sc0[r] : MASKVAL;
      sv[8 + r] = ok1 ? sc1[r] : MASKVAL;
    }
    float tmax = sv[0];
#pragma unroll
    for (int j = 1; j < 16; ++j) tmax = fmaxf(tmax, sv[j]);
    tmax = fmaxf(tmax, __shfl_xor(tmax, 16, 32));
    const float m_new = fmaxf(m_run, tmax);
    const float scale = exp2f((m_run - m_new) * LOG2E_E);

    // P = exp(S - m_new); C-layout of S^T maps directly onto A-layout of P
    BF16Frag pf;
    float tsum = 0.0f;
#pragma unroll
    for (int j = 0; j < 16; ++j) {
      const float p = exp2f((sv[j] - m_new) * LOG2E_E);
      tsum += p;
      pf.u[j] = f2bf(p);
    }
    tsum += __shfl_xor(tsum, 16, 32);
    l_run = l_run * scale + tsum;
    m_run = m_new;

    // rescale accumulators: O rows are query m = r + hi*8 -> broadcast scale
    float rsc[8];
#pragma unroll
    for (int r = 0; r < 8; ++r) rsc[r] = __shfl(scale, r + hi * 8, 32);
#pragma unroll
    for (int c8 = 0; c8 < 8; ++c8) {
#pragma unroll
      for (int r = 0; r < 8; ++r) acc[c8][r] *= rsc[r];
    }

    // ---- O += P * V : 8 d-column tiles, V^T B-fragments from LDS ----
#pragma unroll
    for (int c8 = 0; c8 < 8; ++c8) {
      BF16Frag vf;
      const unsigned short* vp = &Vt[(c8 * 16 + col) * VSTR + hi * 8];
      vf.q[0] = *(const uint4*)(vp);
      vf.q[1] = *(const uint4*)(vp + 16);
      acc[c8] = wmma_bf16(pf.v, vf.v, acc[c8]);
    }
  }

  // ---- write outputs: O rows = q0 + hi*8 + r, dcol = c8*16 + col ----
#pragma unroll
  for (int c8 = 0; c8 < 8; ++c8) {
#pragma unroll
    for (int r = 0; r < 8; ++r) {
      const int tq = q0 + hi * 8 + r;
      outO[(((size_t)b * T_ + tq) * NH + h) * D_ + c8 * 16 + col] = acc[c8][r];
    }
  }
  if (hi == 0) {
    const size_t idx = ((size_t)b * T_ + (q0 + col)) * NH + h;
    outM[idx] = m_run;
    outL[idx] = l_run;
  }
}

extern "C" void kernel_launch(void* const* d_in, const int* in_sizes, int n_in,
                              void* d_out, int out_size, void* d_ws, size_t ws_size,
                              hipStream_t stream) {
  (void)in_sizes; (void)n_in; (void)out_size; (void)d_ws; (void)ws_size;
  const float* Q = (const float*)d_in[0];
  const float* K = (const float*)d_in[1];
  const float* V = (const float*)d_in[2];
  const int* SEG = (const int*)d_in[3];
  // d_in[4] = num_kv_heads (device scalar) -- shapes are compile-time here.

  float* out  = (float*)d_out;
  float* outO = out;
  float* outM = outO + (size_t)B_ * T_ * NH * D_;
  float* outL = outM + (size_t)B_ * T_ * NH;

  dim3 grid(B_ * NH * (T_ / BQ));
  dim3 block(128);
  fa_fwd_kernel<<<grid, block, 0, stream>>>(Q, K, V, SEG, outO, outM, outL);
}
